// SpeechAttentionWithQ_27865747817307
// MI455X (gfx1250) — compile-verified, matrix operand order checked by reading, not run on hardware
//
#include <hip/hip_runtime.h>
#include <hip/hip_bf16.h>
#include <stdint.h>

// ---------------------------------------------------------------------------
// SpeechAttentionWithQ for MI455X (gfx1250, wave32, WMMA + async global->LDS).
// Pipeline: conv_qkv(+q softmax,+mask) -> k column softmax (writes K^T) ->
//           K^T*V (WMMA, async A) -> Q*KV (WMMA, async A) -> conv1d.
// All WMMA fragments (A and B) are assembled with ds_load_b128 pairs.
// ---------------------------------------------------------------------------

#define BATCH 16
#define CCH   128
#define TLEN  16384
#define MIDC  128
#define PATCH 8
#define WN    2048          // wordnum = TLEN / PATCH
#define CP    1024          // MIDC * PATCH

typedef __attribute__((ext_vector_type(16))) _Float16 v16h;
typedef __attribute__((ext_vector_type(8)))  _Float16 v8h;
typedef __attribute__((ext_vector_type(8)))  float    v8f;

union AFrag { v16h v; v8h h[2]; _Float16 e[16]; };
union HFrag { v8h  v; _Float16 e[8];  };
union CFrag { v8f  v; float    e[8];  };

// LDS aperture: generic address low 32 bits are the LDS byte address.
__device__ __forceinline__ uint32_t lds_addr32(const void* p) {
  return (uint32_t)(uintptr_t)p;
}
// One 16-byte async DMA copy global -> LDS (tracked by ASYNCcnt).
__device__ __forceinline__ void async_copy_b128(uint32_t lds, const void* gsrc) {
  asm volatile("global_load_async_to_lds_b128 %0, %1, off"
               :: "v"(lds), "v"((uint64_t)(uintptr_t)gsrc) : "memory");
}
__device__ __forceinline__ void wait_asynccnt0() {
  asm volatile("s_wait_asynccnt 0x0" ::: "memory");
}

// ---------------------------------------------------------------------------
// Kernel 1: grouped 5-tap convs over word axis + q row-softmax + k mask.
// One block per (b, w). Produces Q(f16), V(f16), Kraw(f32, masked).
// Layouts: Q/V/Kraw are [b][w][m*8+p] (cp contiguous).
// ---------------------------------------------------------------------------
__global__ __launch_bounds__(256) void conv_qkv_kernel(
    const float* __restrict__ x, const float* __restrict__ cond,
    const float* __restrict__ mask,
    const float* __restrict__ wQ, const float* __restrict__ bQ,
    const float* __restrict__ wKV, const float* __restrict__ bKV,
    _Float16* __restrict__ Qbuf, float* __restrict__ Kraw,
    _Float16* __restrict__ Vbuf)
{
  const int blk = blockIdx.x;
  const int b = blk >> 11;        // / WN
  const int w = blk & (WN - 1);
  const int tid = threadIdx.x;

  __shared__ float sC[CCH * 40];  // condition slab: 128 ch x 5 words x 8
  __shared__ float sX[CCH * 40];  // x slab
  __shared__ float sQ[CP];
  __shared__ float sRed[256];

  // Load 40 contiguous t-samples (words w-2..w+2) per channel, zero padded.
  const long tbase = (long)(w - 2) * PATCH;
  for (int idx = tid; idx < CCH * 40; idx += 256) {
    const int ci = idx / 40;
    const int o  = idx - ci * 40;
    const long t = tbase + o;
    float cv = 0.f, xv = 0.f;
    if (t >= 0 && t < TLEN) {
      const long g = ((long)b * CCH + ci) * TLEN + t;
      cv = cond[g];
      xv = x[g];
    }
    sC[ci * 40 + o] = cv;
    sX[ci * 40 + o] = xv;
  }
  __syncthreads();

  const int p  = tid & 7;
  const int mb = tid >> 3;   // 0..31

  // ---- Q conv: out channel m in {mb, mb+32, mb+64, mb+96}, group = m>>5 ----
  #pragma unroll
  for (int qi = 0; qi < 4; ++qi) {
    const int m = mb + 32 * qi;
    const int g = m >> 5;
    float acc = bQ[m];
    const float* wp  = wQ + m * 160;            // (m,ci,dc) -> m*160+ci*5+dc
    const float* cp0 = sC + (g * 32) * 40 + p;
    #pragma unroll 4
    for (int ci = 0; ci < 32; ++ci) {
      const float* cc = cp0 + ci * 40;
      const float* ww = wp + ci * 5;
      acc += cc[0]  * ww[0] + cc[8]  * ww[1] + cc[16] * ww[2]
           + cc[24] * ww[3] + cc[32] * ww[4];
    }
    sQ[m * 8 + p] = acc;
  }

  // ---- KV conv: out channel m2 in 0..255, group = m2>>6 ----
  for (int i = 0; i < 8; ++i) {
    const int m2 = mb + 32 * i;
    const int g2 = m2 >> 6;
    float acc = bKV[m2];
    const float* wp  = wKV + m2 * 160;
    const float* xp0 = sX + (g2 * 32) * 40 + p;
    #pragma unroll 4
    for (int ci = 0; ci < 32; ++ci) {
      const float* xx = xp0 + ci * 40;
      const float* ww = wp + ci * 5;
      acc += xx[0]  * ww[0] + xx[8]  * ww[1] + xx[16] * ww[2]
           + xx[24] * ww[3] + xx[32] * ww[4];
    }
    if (m2 < MIDC) {  // K channel: add dropout mask, keep f32 for col-softmax
      acc += mask[(((long)b * MIDC + m2) * WN + w) * PATCH + p];
      Kraw[((long)b * WN + w) * CP + m2 * 8 + p] = acc;
    } else {          // V channel: straight to f16
      const int m = m2 - MIDC;
      Vbuf[((long)b * WN + w) * CP + m * 8 + p] = (_Float16)acc;
    }
  }
  __syncthreads();

  // ---- row softmax of q over cp=1024 ----
  float lmax = -3.4e38f;
  for (int i = tid; i < CP; i += 256) lmax = fmaxf(lmax, sQ[i]);
  sRed[tid] = lmax;
  __syncthreads();
  for (int s = 128; s > 0; s >>= 1) {
    if (tid < s) sRed[tid] = fmaxf(sRed[tid], sRed[tid + s]);
    __syncthreads();
  }
  const float rmax = sRed[0];
  __syncthreads();
  float lsum = 0.f;
  for (int i = tid; i < CP; i += 256) lsum += __expf(sQ[i] - rmax);
  sRed[tid] = lsum;
  __syncthreads();
  for (int s = 128; s > 0; s >>= 1) {
    if (tid < s) sRed[tid] += sRed[tid + s];
    __syncthreads();
  }
  const float inv = 1.0f / sRed[0];
  _Float16* qo = Qbuf + ((long)b * WN + w) * CP;
  for (int i = tid; i < CP; i += 256)
    qo[i] = (_Float16)(__expf(sQ[i] - rmax) * inv);
}

// ---------------------------------------------------------------------------
// Kernel 2: softmax of K along the word axis (2048) per (b, column j).
// Writes K TRANSPOSED: KbufT[b][j][w] with vectorized 16B stores, so the
// K^T*V GEMM can async-DMA its A tiles straight into LDS.
// ---------------------------------------------------------------------------
__global__ __launch_bounds__(256) void ksoftmax_kernel(
    const float* __restrict__ Kraw, _Float16* __restrict__ KbufT)
{
  const int b = blockIdx.y;
  const int j = blockIdx.x * 256 + threadIdx.x;
  const float* kp = Kraw + (long)b * WN * CP + j;
  float m = -3.4e38f;
  for (int w = 0; w < WN; ++w) m = fmaxf(m, kp[(long)w * CP]);
  float s = 0.f;
  for (int w = 0; w < WN; ++w) s += __expf(kp[(long)w * CP] - m);
  const float inv = 1.0f / s;
  _Float16* ko = KbufT + ((long)b * CP + j) * WN;   // row j, contiguous in w
  for (int w = 0; w < WN; w += 8) {
    HFrag h;
    #pragma unroll
    for (int e = 0; e < 8; ++e)
      h.e[e] = (_Float16)(__expf(kp[(long)(w + e) * CP] - m) * inv);
    *(v8h*)(ko + w) = h.v;
  }
}

// ---------------------------------------------------------------------------
// WMMA 16-bit fragment layout (ISA 7.12.2): per lane the 16 halfs are two
// contiguous 8-half chunks at K = kh*8 and K = 16 + kh*8 (kh = lane>>4),
// row = lane&15.  With [row][k] LDS tiles (16B-aligned rows) every fragment
// is exactly two ds_load_b128.
// ---------------------------------------------------------------------------

// ---------------------------------------------------------------------------
// Kernel 3: KV[i][j] = sum_w K[w][i] * V[w][j]   (per batch, 1024x1024, K=2048)
// Block tile 128(i) x 64(j); 8 waves, each 32x32 via 2x2 v_wmma_f32_16x16x32_f16.
// A = KbufT rows (layout-preserving)  -> async global->LDS DMA.
// B = V rows, needed [j][k] in LDS    -> staged transpose (b128 + 8 ds stores).
// ---------------------------------------------------------------------------
__global__ __launch_bounds__(256) void gemm_ktv_kernel(
    const _Float16* __restrict__ KbufT, const _Float16* __restrict__ Vbuf,
    _Float16* __restrict__ KVbuf)
{
  const int b  = blockIdx.z;
  const int i0 = blockIdx.y * 128;
  const int j0 = blockIdx.x * 64;
  const int tid  = threadIdx.x;
  const int lane = tid & 31;
  const int wave = tid >> 5;
  const int wm = wave & 3;      // 4 waves along i (32 rows each)
  const int wn = wave >> 2;     // 2 waves along j (32 cols each)

  __shared__ __align__(16) _Float16 sA[128][40];  // [i][k], 80B rows
  __shared__ __align__(16) _Float16 sBt[64][40];  // [j][k], 80B rows

  v8f acc[2][2] = {};

  const _Float16* KbT = KbufT + (long)b * CP * WN;
  const _Float16* Vb  = Vbuf + (long)b * WN * CP;

  const int lm = lane & 15;
  const int kh = lane >> 4;

  // Async-A: 128x32 halfs = 512 x 16B chunks, 2 per thread.
  const int ai = tid >> 2;                  // 0..63 (+64 for second chunk)
  const int ac = tid & 3;                   // 4 chunks of 8 k's
  const uint32_t ldsA0 = lds_addr32(&sA[ai][ac * 8]);
  const uint32_t ldsA1 = lds_addr32(&sA[ai + 64][ac * 8]);
  // B staging: k = tid>>3 (0..31), 8 j's per chunk.
  const int bk = tid >> 3;
  const int bc = tid & 7;

  for (int wk = 0; wk < WN; wk += 32) {
    __syncthreads();
    // A tile: async DMA rows of K^T (contiguous in w).
    async_copy_b128(ldsA0, KbT + (long)(i0 + ai) * WN + wk + ac * 8);
    async_copy_b128(ldsA1, KbT + (long)(i0 + ai + 64) * WN + wk + ac * 8);

    // B tile: transpose-stage V rows into [j][k].
    {
      HFrag h;
      h.v = *(const v8h*)(Vb + (long)(wk + bk) * CP + j0 + bc * 8);
      #pragma unroll
      for (int e = 0; e < 8; ++e) sBt[bc * 8 + e][bk] = h.e[e];
    }
    if (wk + 32 < WN)
      __builtin_prefetch(Vb + (long)(wk + 32 + bk) * CP + j0 + bc * 8, 0, 0);

    wait_asynccnt0();
    __syncthreads();

    AFrag a[2], bf[2];
    #pragma unroll
    for (int mi = 0; mi < 2; ++mi) {
      const int ib = wm * 32 + mi * 16 + lm;
      a[mi].h[0] = *(const v8h*)&sA[ib][kh * 8];
      a[mi].h[1] = *(const v8h*)&sA[ib][16 + kh * 8];
    }
    #pragma unroll
    for (int ni = 0; ni < 2; ++ni) {
      const int jb = wn * 32 + ni * 16 + lm;
      bf[ni].h[0] = *(const v8h*)&sBt[jb][kh * 8];
      bf[ni].h[1] = *(const v8h*)&sBt[jb][16 + kh * 8];
    }
    #pragma unroll
    for (int mi = 0; mi < 2; ++mi)
      #pragma unroll
      for (int ni = 0; ni < 2; ++ni)
        acc[mi][ni] = __builtin_amdgcn_wmma_f32_16x16x32_f16(
            false, a[mi].v, false, bf[ni].v, (short)0, acc[mi][ni],
            false, false);
  }

  // C/D layout: VGPR r -> M = r (lanes 0..15) / r+8 (lanes 16..31); N = lane&15
  const int Moff = (lane >> 4) * 8;
  const int N = lane & 15;
  _Float16* Cb = KVbuf + (long)b * CP * CP;
  #pragma unroll
  for (int mi = 0; mi < 2; ++mi)
    #pragma unroll
    for (int ni = 0; ni < 2; ++ni) {
      CFrag cf; cf.v = acc[mi][ni];
      const int gi0 = i0 + wm * 32 + mi * 16 + Moff;
      const int gj  = j0 + wn * 32 + ni * 16 + N;
      #pragma unroll
      for (int r = 0; r < 8; ++r)
        Cb[(long)(gi0 + r) * CP + gj] = (_Float16)cf.e[r];
    }
}

// ---------------------------------------------------------------------------
// Kernel 4: Out[m][j] = sum_k Q[m][k] * KV[k][j]  (per batch, 2048x1024, K=1024)
// A = Q rows (layout-preserving) -> async DMA.  B = staged transpose.
// ---------------------------------------------------------------------------
__global__ __launch_bounds__(256) void gemm_qkv_kernel(
    const _Float16* __restrict__ Qbuf, const _Float16* __restrict__ KVbuf,
    float* __restrict__ OutMid)
{
  const int b  = blockIdx.z;
  const int m0 = blockIdx.y * 128;
  const int j0 = blockIdx.x * 64;
  const int tid  = threadIdx.x;
  const int lane = tid & 31;
  const int wave = tid >> 5;
  const int wm = wave & 3;
  const int wn = wave >> 2;

  __shared__ __align__(16) _Float16 sA[128][40];  // [m][k]
  __shared__ __align__(16) _Float16 sBt[64][40];  // [j][k]

  v8f acc[2][2] = {};

  const _Float16* Qb = Qbuf + (long)b * WN * CP;
  const _Float16* Kv = KVbuf + (long)b * CP * CP;

  const int lm = lane & 15;
  const int kh = lane >> 4;

  // Async-A: 128x32 halfs = 512 x 16B chunks, 2 per thread.
  const int am = tid >> 2;                  // 0..63 (+64 for second chunk)
  const int ac = tid & 3;                   // 4 chunks of 8 k's
  const uint32_t ldsA0 = lds_addr32(&sA[am][ac * 8]);
  const uint32_t ldsA1 = lds_addr32(&sA[am + 64][ac * 8]);
  // B staging indices.
  const int bk = tid >> 3;
  const int bc = tid & 7;

  for (int k0 = 0; k0 < CP; k0 += 32) {
    __syncthreads();
    async_copy_b128(ldsA0, Qb + (long)(m0 + am) * CP + k0 + ac * 8);
    async_copy_b128(ldsA1, Qb + (long)(m0 + am + 64) * CP + k0 + ac * 8);
    {
      HFrag h;
      h.v = *(const v8h*)(Kv + (long)(k0 + bk) * CP + j0 + bc * 8);
      #pragma unroll
      for (int e = 0; e < 8; ++e) sBt[bc * 8 + e][bk] = h.e[e];
    }
    if (k0 + 32 < CP)
      __builtin_prefetch(Kv + (long)(k0 + 32 + bk) * CP + j0 + bc * 8, 0, 0);

    wait_asynccnt0();
    __syncthreads();

    AFrag a[2], bf[2];
    #pragma unroll
    for (int mi = 0; mi < 2; ++mi) {
      const int ib = wm * 32 + mi * 16 + lm;
      a[mi].h[0] = *(const v8h*)&sA[ib][kh * 8];
      a[mi].h[1] = *(const v8h*)&sA[ib][16 + kh * 8];
    }
    #pragma unroll
    for (int ni = 0; ni < 2; ++ni) {
      const int jb = wn * 32 + ni * 16 + lm;
      bf[ni].h[0] = *(const v8h*)&sBt[jb][kh * 8];
      bf[ni].h[1] = *(const v8h*)&sBt[jb][16 + kh * 8];
    }
    #pragma unroll
    for (int mi = 0; mi < 2; ++mi)
      #pragma unroll
      for (int ni = 0; ni < 2; ++ni)
        acc[mi][ni] = __builtin_amdgcn_wmma_f32_16x16x32_f16(
            false, a[mi].v, false, bf[ni].v, (short)0, acc[mi][ni],
            false, false);
  }

  const int Moff = (lane >> 4) * 8;
  const int N = lane & 15;
  float* Ob = OutMid + (long)b * WN * CP;
  #pragma unroll
  for (int mi = 0; mi < 2; ++mi)
    #pragma unroll
    for (int ni = 0; ni < 2; ++ni) {
      CFrag cf; cf.v = acc[mi][ni];
      const int gm0 = m0 + wm * 32 + mi * 16 + Moff;
      const int gj  = j0 + wn * 32 + ni * 16 + N;
      #pragma unroll
      for (int r = 0; r < 8; ++r)
        Ob[(long)(gm0 + r) * CP + gj] = cf.e[r];
    }
}

// ---------------------------------------------------------------------------
// Kernel 5: 3-tap conv1d over t: out[b][co][t] = bOut[co]
//           + sum_ci sum_d mid[b][ci][t+d-1] * wOut[co][ci][d]
// mid(b,ci,t) lives at OutMid[b][t>>3][ci*8 + (t&7)].
// ---------------------------------------------------------------------------
__global__ __launch_bounds__(256) void conv1d_out_kernel(
    const float* __restrict__ OutMid, const float* __restrict__ wOut,
    const float* __restrict__ bOut, float* __restrict__ out)
{
  const int b  = blockIdx.y;
  const int t0 = blockIdx.x * 64;
  const int tid = threadIdx.x;

  __shared__ float sM[CCH * 66];   // [ci][lt], lt = t - (t0 - 1)

  for (int i = tid; i < CCH * 66; i += 256) sM[i] = 0.f;
  __syncthreads();

  const int wlo = (t0 - 1) >> 3;   // arithmetic shift: t0==0 -> -1 (skipped)
  const int whi = (t0 + 64) >> 3;
  for (int w = wlo; w <= whi; ++w) {
    if (w < 0 || w >= WN) continue;
    const float* src = OutMid + ((long)b * WN + w) * CP;
    for (int j = tid; j < CP; j += 256) {
      const int ci = j >> 3, p = j & 7;
      const int lt = w * 8 + p - (t0 - 1);
      if (lt >= 0 && lt < 66) sM[ci * 66 + lt] = src[j];
    }
  }
  __syncthreads();

  for (int it = 0; it < 32; ++it) {
    const int o   = it * 256 + tid;   // 0..8191 = 128 co x 64 t
    const int lt8 = o & 63;
    const int co  = o >> 6;
    const float* wp = wOut + co * (CCH * 3);
    float acc = bOut[co];
    const int ltc = lt8 + 1;
    #pragma unroll 4
    for (int ci = 0; ci < CCH; ++ci) {
      const float* s = sM + ci * 66 + ltc;
      acc += s[-1] * wp[ci * 3 + 0] + s[0] * wp[ci * 3 + 1]
           + s[1] * wp[ci * 3 + 2];
    }
    out[((long)b * CCH + co) * TLEN + t0 + lt8] = acc;
  }
}

// ---------------------------------------------------------------------------
// Workspace layout (bytes):
//   Qbuf  f16 : [0,          64 MB)
//   KbufT f16 : [64 MB,     128 MB)   (K softmaxed, transposed [b][j][w])
//   Vbuf  f16 : [128 MB,    192 MB)
//   KV    f16 : [192 MB,    224 MB)
//   Kraw  f32 : [224 MB,    352 MB)  -- reused later as OutMid f32 (disjoint
//                                       lifetimes: Kraw dead after ksoftmax)
// ---------------------------------------------------------------------------
extern "C" void kernel_launch(void* const* d_in, const int* in_sizes, int n_in,
                              void* d_out, int out_size, void* d_ws, size_t ws_size,
                              hipStream_t stream) {
  (void)in_sizes; (void)n_in; (void)out_size; (void)ws_size;

  const float* x    = (const float*)d_in[0];
  const float* cond = (const float*)d_in[1];
  const float* mask = (const float*)d_in[2];
  const float* wQ   = (const float*)d_in[3];
  const float* bQ   = (const float*)d_in[4];
  const float* wKV  = (const float*)d_in[5];
  const float* bKV  = (const float*)d_in[6];
  const float* wOut = (const float*)d_in[7];
  const float* bOut = (const float*)d_in[8];
  float* out = (float*)d_out;

  char* ws = (char*)d_ws;
  const size_t MB = 1024ull * 1024ull;
  _Float16* Qbuf  = (_Float16*)(ws + 0);
  _Float16* KbufT = (_Float16*)(ws + 64 * MB);
  _Float16* Vbuf  = (_Float16*)(ws + 128 * MB);
  _Float16* KVbuf = (_Float16*)(ws + 192 * MB);
  float*    Kraw  = (float*)(ws + 224 * MB);
  float*    OutMid = Kraw;   // reuse: Kraw consumed before OutMid written

  conv_qkv_kernel<<<BATCH * WN, 256, 0, stream>>>(
      x, cond, mask, wQ, bQ, wKV, bKV, Qbuf, Kraw, Vbuf);

  ksoftmax_kernel<<<dim3(CP / 256, BATCH), 256, 0, stream>>>(Kraw, KbufT);

  gemm_ktv_kernel<<<dim3(CP / 64, CP / 128, BATCH), 256, 0, stream>>>(
      KbufT, Vbuf, KVbuf);

  gemm_qkv_kernel<<<dim3(CP / 64, WN / 128, BATCH), 256, 0, stream>>>(
      Qbuf, KVbuf, OutMid);

  conv1d_out_kernel<<<dim3(TLEN / 64, BATCH), 256, 0, stream>>>(
      OutMid, wOut, bOut, out);
}
